// CrossAttention_38766374814271
// MI455X (gfx1250) — matrix operand order repaired and checked by reading, not compiled, not run on hardware
//
#include <hip/hip_runtime.h>
#include <hip/hip_bf16.h>
#include <stdint.h>

typedef _Float16 half_t;
typedef __attribute__((ext_vector_type(16))) _Float16 v16h;
typedef __attribute__((ext_vector_type(8)))  _Float16 v8h;
typedef __attribute__((ext_vector_type(4)))  _Float16 v4h;
typedef __attribute__((ext_vector_type(8)))  float    v8f;
typedef __attribute__((ext_vector_type(4)))  float    v4f;

#define B_ 2
#define Q_ 1024
#define N_ 4096
#define C_ 768
#define H_ 12
#define D_ 64
#define SCALE_ 0.125f   // 1/sqrt(64)

// ---------------------------------------------------------------------------
// gfx1250 async global->LDS copy (ASYNCcnt-tracked), with portable fallback.
// Builtin signature (from hipcc diagnostic): params are pointers to
// 'int __attribute__((vector_size(16)))' in AS1 (global) / AS3 (LDS).
// ---------------------------------------------------------------------------
#if __has_builtin(__builtin_amdgcn_global_load_async_to_lds_b128) && \
    __has_builtin(__builtin_amdgcn_s_wait_asynccnt)
#define USE_ASYNC 1
#else
#define USE_ASYNC 0
#endif

typedef int vint4 __attribute__((vector_size(16)));
typedef __attribute__((address_space(1))) vint4* as1_v4i_ptr;
typedef __attribute__((address_space(3))) vint4* as3_v4i_ptr;

__device__ __forceinline__ void cp16(half_t* lds, const half_t* g) {
#if USE_ASYNC
  // generic->as(3): low 32 bits of a generic LDS address are the LDS offset.
  __builtin_amdgcn_global_load_async_to_lds_b128(
      (as1_v4i_ptr)(uintptr_t)g,
      (as3_v4i_ptr)(uint32_t)(uintptr_t)lds,
      0, 0);
#else
  *(v8h*)lds = *(const v8h*)g;
#endif
}

template <int N> __device__ __forceinline__ void wait_async() {
#if USE_ASYNC
  __builtin_amdgcn_s_wait_asynccnt(N);
#endif
}

// ---------------------------------------------------------------------------
// WMMA helper: D = A(16x32 f16) * B(32x16 f16) + C(16x16 f32)
// ---------------------------------------------------------------------------
__device__ __forceinline__ v8f wmma16x16x32(v16h a, v16h b, v8f c) {
  return __builtin_amdgcn_wmma_f32_16x16x32_f16(
      false, a, false, b, (short)0, c, false, false);
}

// A fragment: 16x32 tile, row-major [row][k], stride ld halves.
// lanes 0-15 -> M=lane, halves K0..7,K16..23; lanes 16-31 -> K8..15,K24..31
__device__ __forceinline__ v16h load_frag_a(const half_t* p, int ld) {
  int lane = threadIdx.x & 31;
  int r  = lane & 15;
  int ko = (lane >> 4) << 3;
  const half_t* q = p + r * ld + ko;
  v16h f;
#pragma unroll
  for (int i = 0; i < 8; ++i) { f[i] = q[i]; f[i + 8] = q[i + 16]; }
  return f;
}

// B fragment: Bt[col][k] row-major (column-major B), stride ld.
// lanes 0-15 -> col=lane, K0..7,K16..23; lanes 16-31 -> col=lane-16, K8..15,K24..31
__device__ __forceinline__ v16h load_frag_b(const half_t* p, int ld) {
  int lane = threadIdx.x & 31;
  int c  = lane & 15;
  int ko = (lane >> 4) << 3;
  const half_t* q = p + c * ld + ko;
  v16h f;
#pragma unroll
  for (int i = 0; i < 8; ++i) { f[i] = q[i]; f[i + 8] = q[i + 16]; }
  return f;
}

// ---------------------------------------------------------------------------
// One-time fp32 -> fp16 conversion (elementwise).
// ---------------------------------------------------------------------------
__global__ __launch_bounds__(256) void cvt_f16_kernel(
    const float* __restrict__ in, half_t* __restrict__ out, int n) {
  int i = (blockIdx.x * 256 + threadIdx.x) * 4;
  if (i < n) {
    v4f x = *(const v4f*)&in[i];
    v4h y;
#pragma unroll
    for (int j = 0; j < 4; ++j) y[j] = (half_t)x[j];
    *(v4h*)&out[i] = y;
  }
}

// One-time W[768k][768n] fp32 -> WT[n][k] fp16 (tiled transpose).
__global__ __launch_bounds__(256) void wtrans_kernel(
    const float* __restrict__ W, half_t* __restrict__ WT) {
  __shared__ half_t T[32][33];
  int k0 = (blockIdx.x % 24) * 32;
  int n0 = (blockIdx.x / 24) * 32;
  int c = threadIdx.x & 31, r8 = threadIdx.x >> 5;
#pragma unroll
  for (int rr = 0; rr < 32; rr += 8)
    T[r8 + rr][c] = (half_t)W[(size_t)(k0 + r8 + rr) * C_ + n0 + c];
  __syncthreads();
#pragma unroll
  for (int rr = 0; rr < 32; rr += 8)
    WT[(size_t)(n0 + r8 + rr) * C_ + k0 + c] = T[c][r8 + rr];
}

// ---------------------------------------------------------------------------
// Projection GEMM: Y[M,768] = A16[M,768] @ W + bias, W given transposed fp16.
// Block tile 64x128, 8 waves (4 row x 2 col), K-step 32, double-buffered
// async global->LDS staging.
//   MODE 0: Qh[B,H,Q,D] fp16, * SCALE   MODE 1: Kh[B,H,N,D] fp16
//   MODE 2: Vt[B,H,D,N] fp16            MODE 3: fp32 out[M,768]
// ---------------------------------------------------------------------------
template <int MODE>
__global__ __launch_bounds__(256) void gemm_proj_kernel(
    const half_t* __restrict__ A, const half_t* __restrict__ WT,
    const float* __restrict__ bias,
    half_t* __restrict__ outh, float* __restrict__ outf, int seqlen)
{
  __shared__ half_t Xs[2][64][40];
  __shared__ half_t Ws[2][128][40];

  const int tid  = threadIdx.x;
  const int wave = tid >> 5;
  const int lane = tid & 31;
  const int wr = wave & 3;
  const int wc = wave >> 2;
  const int row0 = blockIdx.x * 64;
  const int col0 = blockIdx.y * 128;

  const int xr = tid >> 2, xs = (tid & 3) << 3;   // X: 1x b128 / thread
  const int wrr = tid >> 1, wss = (tid & 1) << 4; // W: 2x b128 / thread

  auto stage = [&](int kb, int buf) {
    int k0 = kb * 32;
    cp16(&Xs[buf][xr][xs],        &A [(size_t)(row0 + xr)  * C_ + k0 + xs]);
    cp16(&Ws[buf][wrr][wss],      &WT[(size_t)(col0 + wrr) * C_ + k0 + wss]);
    cp16(&Ws[buf][wrr][wss + 8],  &WT[(size_t)(col0 + wrr) * C_ + k0 + wss + 8]);
  };

  v8f acc[4];
#pragma unroll
  for (int t = 0; t < 4; ++t)
#pragma unroll
    for (int j = 0; j < 8; ++j) acc[t][j] = 0.0f;

  stage(0, 0);
  for (int kb = 0; kb < 24; ++kb) {
    const int buf = kb & 1;
    if (kb + 1 < 24) { stage(kb + 1, buf ^ 1); wait_async<3>(); }
    else             { wait_async<0>(); }
    __syncthreads();

    v16h a = load_frag_a(&Xs[buf][wr * 16][0], 40);
#pragma unroll
    for (int bt = 0; bt < 4; ++bt) {
      v16h b = load_frag_b(&Ws[buf][wc * 64 + bt * 16][0], 40);
      acc[bt] = wmma16x16x32(a, b, acc[bt]);
    }
    __syncthreads();
  }

  const int hi8 = (lane >> 4) << 3;
#pragma unroll
  for (int bt = 0; bt < 4; ++bt) {
    int n = col0 + wc * 64 + bt * 16 + (lane & 15);
    float bv = bias[n];
    int h = n >> 6, d = n & 63;
#pragma unroll
    for (int vr = 0; vr < 8; ++vr) {
      int m = row0 + wr * 16 + vr + hi8;
      float v = acc[bt][vr] + bv;
      int bidx = m / seqlen, s = m % seqlen;
      if constexpr (MODE == 0) {
        outh[((size_t)(bidx * H_ + h) * Q_ + s) * D_ + d] = (half_t)(v * SCALE_);
      } else if constexpr (MODE == 1) {
        outh[((size_t)(bidx * H_ + h) * N_ + s) * D_ + d] = (half_t)v;
      } else if constexpr (MODE == 2) {
        outh[((size_t)(bidx * H_ + h) * D_ + d) * N_ + s] = (half_t)v;
      } else {
        outf[(size_t)m * C_ + n] = v;
      }
    }
  }
}

// ---------------------------------------------------------------------------
// Flash attention: 8 waves x 16 q-rows = 128 q rows / block. Double-buffered
// async staging of 64-key K/V chunks; online softmax; bias as WMMA C-init.
// ---------------------------------------------------------------------------
__global__ __launch_bounds__(256) void flash_attn_kernel(
    const half_t* __restrict__ Qh, const half_t* __restrict__ Kh,
    const half_t* __restrict__ Vt, const float* __restrict__ relb,
    half_t* __restrict__ Oh)
{
  __shared__ half_t Kc[2][64][72];
  __shared__ half_t Vc[2][64][72];
  __shared__ half_t Pc[8][16][72];
  __shared__ float  Bq[128][32];

  const int tid  = threadIdx.x;
  const int wave = tid >> 5;
  const int lane = tid & 31;
  const int bh = blockIdx.y;
  const int b  = bh / H_;
  const int h  = bh % H_;
  const int q0b = blockIdx.x * 128;
  const int q0w = q0b + wave * 16;
  const int hi8 = (lane >> 4) << 3;

  const half_t* qbase = Qh + ((size_t)bh * Q_ + q0w) * D_;
  const half_t* kbase = Kh + (size_t)bh * N_ * D_;
  const half_t* vbase = Vt + (size_t)bh * D_ * N_;
  const float*  bb    = relb + h * 32 * 32;

  // granule mapping for async staging: 512 b128 granules per 64x64 chunk
  const int r0 = tid >> 3,         s0 = (tid & 7) << 3;
  const int r1 = (tid + 256) >> 3, s1 = ((tid + 256) & 7) << 3;

  auto stage_kv = [&](int nc, int buf) {
    cp16(&Kc[buf][r0][s0], &kbase[(size_t)(nc + r0) * D_ + s0]);
    cp16(&Kc[buf][r1][s1], &kbase[(size_t)(nc + r1) * D_ + s1]);
    cp16(&Vc[buf][r0][s0], &vbase[(size_t)r0 * N_ + nc + s0]);
    cp16(&Vc[buf][r1][s1], &vbase[(size_t)r1 * N_ + nc + s1]);
  };

  // pre-lerp bias along q (align_corners=True)
  for (int i = tid; i < 128 * 32; i += 256) {
    int qi = i >> 5, n32 = i & 31;
    float pos = (float)(q0b + qi) * (31.0f / 1023.0f);
    float fl = floorf(pos);
    int lo = (int)fl; if (lo > 31) lo = 31; if (lo < 0) lo = 0;
    int hi = lo + 1;  if (hi > 31) hi = 31;
    float w = pos - fl;
    Bq[qi][n32] = bb[lo * 32 + n32] * (1.0f - w) + bb[hi * 32 + n32] * w;
  }

  v16h aq0 = load_frag_a(qbase,      D_);
  v16h aq1 = load_frag_a(qbase + 32, D_);

  v8f o[4];
  float mrow[8], lrow[8];
#pragma unroll
  for (int dt = 0; dt < 4; ++dt)
#pragma unroll
    for (int j = 0; j < 8; ++j) o[dt][j] = 0.0f;
#pragma unroll
  for (int vr = 0; vr < 8; ++vr) { mrow[vr] = -1e30f; lrow[vr] = 0.0f; }

  int qloc[8];
#pragma unroll
  for (int vr = 0; vr < 8; ++vr) qloc[vr] = wave * 16 + vr + hi8;

  stage_kv(0, 0);
  for (int it = 0; it < N_ / 64; ++it) {
    const int nc = it * 64;
    const int buf = it & 1;
    if (it + 1 < N_ / 64) { stage_kv(nc + 64, buf ^ 1); wait_async<4>(); }
    else                  { wait_async<0>(); }
    __syncthreads();

    // S init with bilinear bias (lerp along n here)
    v8f s[4];
#pragma unroll
    for (int nt = 0; nt < 4; ++nt) {
      int ng = nc + nt * 16 + (lane & 15);
      float posn = (float)ng * (31.0f / 4095.0f);
      float fln = floorf(posn);
      int ln = (int)fln; if (ln > 31) ln = 31; if (ln < 0) ln = 0;
      int hn = ln + 1;   if (hn > 31) hn = 31;
      float wn = posn - fln;
#pragma unroll
      for (int vr = 0; vr < 8; ++vr)
        s[nt][vr] = Bq[qloc[vr]][ln] * (1.0f - wn) + Bq[qloc[vr]][hn] * wn;
    }
    // S += Q * K^T
#pragma unroll
    for (int nt = 0; nt < 4; ++nt) {
      v16h bk0 = load_frag_b(&Kc[buf][nt * 16][0],  72);
      v16h bk1 = load_frag_b(&Kc[buf][nt * 16][32], 72);
      s[nt] = wmma16x16x32(aq0, bk0, s[nt]);
      s[nt] = wmma16x16x32(aq1, bk1, s[nt]);
    }

    // online softmax (row = one VGPR index across a 16-lane group)
#pragma unroll
    for (int vr = 0; vr < 8; ++vr) {
      float mx = fmaxf(fmaxf(s[0][vr], s[1][vr]), fmaxf(s[2][vr], s[3][vr]));
#pragma unroll
      for (int mk = 1; mk <= 8; mk <<= 1)
        mx = fmaxf(mx, __shfl_xor(mx, mk, 32));
      float mnew  = fmaxf(mrow[vr], mx);
      float alpha = __expf(mrow[vr] - mnew);
      mrow[vr] = mnew;
      float rs = 0.0f;
#pragma unroll
      for (int nt = 0; nt < 4; ++nt) {
        float p = __expf(s[nt][vr] - mnew);
        s[nt][vr] = p;
        rs += p;
      }
#pragma unroll
      for (int mk = 1; mk <= 8; mk <<= 1)
        rs += __shfl_xor(rs, mk, 32);
      lrow[vr] = lrow[vr] * alpha + rs;
#pragma unroll
      for (int dt = 0; dt < 4; ++dt) o[dt][vr] *= alpha;
    }

    // relayout P through LDS: C/D layout -> A layout (fp16)
#pragma unroll
    for (int nt = 0; nt < 4; ++nt)
#pragma unroll
      for (int vr = 0; vr < 8; ++vr)
        Pc[wave][vr + hi8][nt * 16 + (lane & 15)] = (half_t)s[nt][vr];

    // O += P * V
#pragma unroll
    for (int c = 0; c < 2; ++c) {
      v16h ap = load_frag_a(&Pc[wave][0][c * 32], 72);
#pragma unroll
      for (int dt = 0; dt < 4; ++dt) {
        v16h bv = load_frag_b(&Vc[buf][dt * 16][c * 32], 72);
        o[dt] = wmma16x16x32(ap, bv, o[dt]);
      }
    }
    __syncthreads();
  }

  // finalize: divide by row sums, store fp16 [b][q][h*64+d]
#pragma unroll
  for (int vr = 0; vr < 8; ++vr) {
    float inv = 1.0f / lrow[vr];
#pragma unroll
    for (int dt = 0; dt < 4; ++dt) o[dt][vr] *= inv;
  }
#pragma unroll
  for (int dt = 0; dt < 4; ++dt) {
    int d = dt * 16 + (lane & 15);
#pragma unroll
    for (int vr = 0; vr < 8; ++vr) {
      int q = q0w + vr + hi8;
      Oh[((size_t)(b * Q_ + q)) * C_ + h * D_ + d] = (half_t)o[dt][vr];
    }
  }
}

// ---------------------------------------------------------------------------
extern "C" void kernel_launch(void* const* d_in, const int* in_sizes, int n_in,
                              void* d_out, int out_size, void* d_ws, size_t ws_size,
                              hipStream_t stream) {
  (void)in_sizes; (void)n_in; (void)out_size; (void)ws_size;
  const float* query  = (const float*)d_in[0];
  const float* keyval = (const float*)d_in[1];
  const float* Wq = (const float*)d_in[2];
  const float* bq = (const float*)d_in[3];
  const float* Wk = (const float*)d_in[4];
  const float* bk = (const float*)d_in[5];
  const float* Wv = (const float*)d_in[6];
  const float* bv = (const float*)d_in[7];
  const float* Wo = (const float*)d_in[8];
  const float* bo = (const float*)d_in[9];
  const float* relb = (const float*)d_in[10];

  half_t* p = (half_t*)d_ws;
  half_t* Xq  = p; p += (size_t)B_ * Q_ * C_;          // query fp16
  half_t* Xkv = p; p += (size_t)B_ * N_ * C_;          // key_value fp16
  half_t* WTq = p; p += (size_t)C_ * C_;               // W^T fp16
  half_t* WTk = p; p += (size_t)C_ * C_;
  half_t* WTv = p; p += (size_t)C_ * C_;
  half_t* WTo = p; p += (size_t)C_ * C_;
  half_t* Qh  = p; p += (size_t)B_ * H_ * Q_ * D_;     // [B,H,Q,D] pre-scaled
  half_t* Kh  = p; p += (size_t)B_ * H_ * N_ * D_;     // [B,H,N,D]
  half_t* Vt  = p; p += (size_t)B_ * H_ * N_ * D_;     // [B,H,D,N]
  half_t* Oh  = p;                                     // [B*Q, C]

  dim3 blk(256);
  const int nq = B_ * Q_ * C_, nkv = B_ * N_ * C_;
  cvt_f16_kernel<<<nq  / 1024, blk, 0, stream>>>(query,  Xq,  nq);
  cvt_f16_kernel<<<nkv / 1024, blk, 0, stream>>>(keyval, Xkv, nkv);
  wtrans_kernel<<<576, blk, 0, stream>>>(Wq, WTq);
  wtrans_kernel<<<576, blk, 0, stream>>>(Wk, WTk);
  wtrans_kernel<<<576, blk, 0, stream>>>(Wv, WTv);
  wtrans_kernel<<<576, blk, 0, stream>>>(Wo, WTo);

  gemm_proj_kernel<0><<<dim3((B_ * Q_) / 64, C_ / 128), blk, 0, stream>>>(
      Xq, WTq, bq, Qh, nullptr, Q_);
  gemm_proj_kernel<1><<<dim3((B_ * N_) / 64, C_ / 128), blk, 0, stream>>>(
      Xkv, WTk, bk, Kh, nullptr, N_);
  gemm_proj_kernel<2><<<dim3((B_ * N_) / 64, C_ / 128), blk, 0, stream>>>(
      Xkv, WTv, bv, Vt, nullptr, N_);
  flash_attn_kernel<<<dim3(Q_ / 128, B_ * H_), blk, 0, stream>>>(
      Qh, Kh, Vt, relb, Oh);
  gemm_proj_kernel<3><<<dim3((B_ * Q_) / 64, C_ / 128), blk, 0, stream>>>(
      Oh, WTo, bo, nullptr, (float*)d_out, Q_);
}
